// MixtureOfExperts_64407329570976
// MI455X (gfx1250) — matrix operand order repaired and checked by reading, not compiled, not run on hardware
//
#include <hip/hip_runtime.h>
#include <hip/hip_bf16.h>
#include <math.h>

#define BATCH 16384
#define DIM   2048
#define NEXP  16
#define NCLS  64
#define KWAVES 4
#define KSLICE (DIM / KWAVES)   // 512 -> 16 WMMA k-steps per wave

typedef __attribute__((ext_vector_type(16))) _Float16 v16h;
typedef __attribute__((ext_vector_type(8)))  float    v8f;

union HVec { v16h v; _Float16 h[16]; uint4 q[2]; };

// A-operand (16x32 f16, M x K) for v_wmma_f32_16x16x32_f16, converted from f32
// on the fly. Per ISA 7.12.2: lane L holds row M=L&15; lanes 0-15 carry
// K = {0..7, 16..23}, lanes 16-31 carry K = {8..15, 24..31}.
__device__ __forceinline__ HVec load_a_f32(const float* __restrict__ row, int k0, int lane) {
  const int kb = (lane & 16) ? 8 : 0;
  const float* p = row + k0 + kb;
  float t[16];
  *reinterpret_cast<float4*>(t +  0) = *reinterpret_cast<const float4*>(p);
  *reinterpret_cast<float4*>(t +  4) = *reinterpret_cast<const float4*>(p + 4);
  *reinterpret_cast<float4*>(t +  8) = *reinterpret_cast<const float4*>(p + 16);
  *reinterpret_cast<float4*>(t + 12) = *reinterpret_cast<const float4*>(p + 20);
  HVec r;
#pragma unroll
  for (int j = 0; j < 16; ++j) r.h[j] = (_Float16)t[j];
  return r;
}

// Same A fragment, loaded directly from the staged f16 copy of the features
// (two 16B loads, no conversion VALU).
__device__ __forceinline__ v16h load_a_f16(const _Float16* __restrict__ row, int k0, int lane) {
  const int kb = (lane & 16) ? 8 : 0;
  const _Float16* p = row + k0 + kb;
  HVec r;
  r.q[0] = *reinterpret_cast<const uint4*>(p);
  r.q[1] = *reinterpret_cast<const uint4*>(p + 16);
  return r.v;
}

// B-operand (32x16 f16, K x N): lane L holds column N=L&15; lanes 0-15 carry
// K = {0..7, 16..23}, lanes 16-31 carry K = {8..15, 24..31} (same as A) from the
// weight row (weights stored [N][K] row-major, i.e. B = W^T).
__device__ __forceinline__ v16h load_b_f16(const _Float16* __restrict__ p) {
  HVec r;
  r.q[0] = *reinterpret_cast<const uint4*>(p);
  r.q[1] = *reinterpret_cast<const uint4*>(p + 16);
  return r.v;
}

__global__ void moe_zero_counts(int* __restrict__ counts) {
  if (threadIdx.x < NEXP) counts[threadIdx.x] = 0;
}

__global__ void moe_cvt_f16(const float* __restrict__ in, _Float16* __restrict__ out, int n4) {
  int i = blockIdx.x * blockDim.x + threadIdx.x;
  if (i >= n4) return;
  float4 v = reinterpret_cast<const float4*>(in)[i];
  union { _Float16 h[4]; uint2 u; } pk;
  pk.h[0] = (_Float16)v.x; pk.h[1] = (_Float16)v.y;
  pk.h[2] = (_Float16)v.z; pk.h[3] = (_Float16)v.w;
  reinterpret_cast<uint2*>(out)[i] = pk.u;
}

// Phase 1: coarse router GEMM. 4 waves per block, each wave accumulates a
// K-slice of 512 into a 16x16 WMMA accumulator; partials reduced through LDS.
// Also optionally stages the f16-converted features for phase 2, does argmax
// routing and bins sample indices into per-expert lists.
template <bool STAGED>
__global__ void __launch_bounds__(32 * KWAVES) moe_coarse(
    const float* __restrict__ feat, const _Float16* __restrict__ cw,
    const float* __restrict__ cb, float* __restrict__ out_coarse,
    float* __restrict__ out_eid, int* __restrict__ counts,
    int* __restrict__ list, _Float16* __restrict__ feat_h) {
  __shared__ float part[KWAVES * 16 * NEXP];  // 4 KB
  const int tid  = threadIdx.x;
  const int lane = tid & 31;
  const int w    = tid >> 5;
  const int m0   = blockIdx.x * 16;
  const int n    = lane & 15;                 // C column / A row / B column
  const int kb   = (lane & 16) ? 8 : 0;
  const size_t rowi = (size_t)(m0 + n) * DIM;
  const float* arow = feat + rowi;
  const _Float16* brow = cw + (size_t)n * DIM + ((lane >> 4) * 8);

  v8f acc = {};
  const int k_end = w * KSLICE + KSLICE;
  for (int k0 = w * KSLICE; k0 < k_end; k0 += 32) {
    HVec a = load_a_f32(arow, k0, lane);
    if (STAGED) {
      *reinterpret_cast<uint4*>(feat_h + rowi + k0 + kb)      = a.q[0];
      *reinterpret_cast<uint4*>(feat_h + rowi + k0 + kb + 16) = a.q[1];
    }
    v16h b = load_b_f16(brow + k0);
    acc = __builtin_amdgcn_wmma_f32_16x16x32_f16(false, a.v, false, b, (short)0, acc, false, false);
  }

  const int mb = (lane & 16) ? 8 : 0;
#pragma unroll
  for (int i = 0; i < 8; ++i) part[w * 256 + (mb + i) * NEXP + n] = acc[i];
  __syncthreads();

  if (w == 0) {
    const float bias = cb[n];
#pragma unroll
    for (int i = 0; i < 8; ++i) {
      const int idx = (mb + i) * NEXP + n;
      float v = part[idx] + part[256 + idx] + part[512 + idx] + part[768 + idx] + bias;
      out_coarse[(size_t)(m0 + mb + i) * NEXP + n] = v;
      part[idx] = v;
    }
  }
  __syncthreads();

  if (tid < 16) {
    float best = part[tid * NEXP];
    int be = 0;
#pragma unroll
    for (int e = 1; e < NEXP; ++e) {
      float v = part[tid * NEXP + e];
      if (v > best) { best = v; be = e; }
    }
    out_eid[m0 + tid] = (float)be;
    int pos = atomicAdd(&counts[be], 1);
    list[be * BATCH + pos] = m0 + tid;
  }
}

// Phase 2: gathered same-expert 16x64 head GEMM. 4 waves split K; each wave
// holds 4 WMMA accumulators (16x16 N-tiles). LDS reduction, then bias +
// softmax + argmax + global class id.
template <bool STAGED>
__global__ void __launch_bounds__(32 * KWAVES) moe_expert(
    const float* __restrict__ feat, const _Float16* __restrict__ feat_h,
    const _Float16* __restrict__ ew, const float* __restrict__ eb,
    const int* __restrict__ counts, const int* __restrict__ list,
    float* __restrict__ out_local, float* __restrict__ out_glob) {
  const int e = blockIdx.y;
  const int t = blockIdx.x;
  const int cnt = counts[e];
  if (t * 16 >= cnt) return;

  __shared__ int   smp[16];
  __shared__ float part[KWAVES * 16 * NCLS];  // 16 KB
  const int tid  = threadIdx.x;
  const int lane = tid & 31;
  const int w    = tid >> 5;
  if (tid < 16) {
    int idx = t * 16 + tid;
    smp[tid] = list[e * BATCH + ((idx < cnt) ? idx : (cnt - 1))];  // pad with valid sample
  }
  __syncthreads();

  const int n    = lane & 15;
  const int kb16 = (lane >> 4) * 8;
  const size_t rowi = (size_t)smp[lane & 15] * DIM;
  const _Float16* wb  = ew + (size_t)e * NCLS * DIM;
  const _Float16* b0p = wb + (size_t)(n +  0) * DIM + kb16;
  const _Float16* b1p = wb + (size_t)(n + 16) * DIM + kb16;
  const _Float16* b2p = wb + (size_t)(n + 32) * DIM + kb16;
  const _Float16* b3p = wb + (size_t)(n + 48) * DIM + kb16;

  v8f c0 = {}, c1 = {}, c2 = {}, c3 = {};
  const int k_end = w * KSLICE + KSLICE;
  for (int k0 = w * KSLICE; k0 < k_end; k0 += 32) {
    v16h a;
    if (STAGED) a = load_a_f16(feat_h + rowi, k0, lane);
    else        a = load_a_f32(feat + rowi, k0, lane).v;
    c0 = __builtin_amdgcn_wmma_f32_16x16x32_f16(false, a, false, load_b_f16(b0p + k0), (short)0, c0, false, false);
    c1 = __builtin_amdgcn_wmma_f32_16x16x32_f16(false, a, false, load_b_f16(b1p + k0), (short)0, c1, false, false);
    c2 = __builtin_amdgcn_wmma_f32_16x16x32_f16(false, a, false, load_b_f16(b2p + k0), (short)0, c2, false, false);
    c3 = __builtin_amdgcn_wmma_f32_16x16x32_f16(false, a, false, load_b_f16(b3p + k0), (short)0, c3, false, false);
  }

  const int mb = (lane & 16) ? 8 : 0;
#pragma unroll
  for (int i = 0; i < 8; ++i) {
    const int base = w * (16 * NCLS) + (mb + i) * NCLS;
    part[base + n +  0] = c0[i];
    part[base + n + 16] = c1[i];
    part[base + n + 32] = c2[i];
    part[base + n + 48] = c3[i];
  }
  __syncthreads();

  // Reduce 4 wave-partials + bias; j*128+tid keeps each wave's LDS accesses
  // contiguous (bank-conflict free).
#pragma unroll
  for (int j = 0; j < 8; ++j) {
    const int idx = j * 128 + tid;  // 0..1023 over 16x64 tile
    float v = part[idx] + part[1024 + idx] + part[2048 + idx] + part[3072 + idx]
            + eb[e * NCLS + (idx & 63)];
    part[idx] = v;
  }
  __syncthreads();

  if (tid < 16 && (t * 16 + tid) < cnt) {
    const int b = smp[tid];
    float* yr = &part[tid * NCLS];
    float mx = yr[0];
    int bi = 0;
#pragma unroll
    for (int c = 1; c < NCLS; ++c) {
      float v = yr[c];
      if (v > mx) { mx = v; bi = c; }
    }
    float s = 0.f;
#pragma unroll
    for (int c = 0; c < NCLS; ++c) {
      float ex = __expf(yr[c] - mx);
      yr[c] = ex;
      s += ex;
    }
    const float inv = 1.f / s;
    float* lp = out_local + (size_t)b * NCLS;
#pragma unroll
    for (int c = 0; c < NCLS; c += 4) {
      float4 v;
      v.x = yr[c + 0] * inv; v.y = yr[c + 1] * inv;
      v.z = yr[c + 2] * inv; v.w = yr[c + 3] * inv;
      *reinterpret_cast<float4*>(lp + c) = v;
    }
    out_glob[b] = (float)(bi + e * NCLS);
  }
}

extern "C" void kernel_launch(void* const* d_in, const int* in_sizes, int n_in,
                              void* d_out, int out_size, void* d_ws, size_t ws_size,
                              hipStream_t stream) {
  const float* feat     = (const float*)d_in[0];
  const float* coarse_w = (const float*)d_in[1];
  const float* coarse_b = (const float*)d_in[2];
  const float* expert_w = (const float*)d_in[3];
  const float* expert_b = (const float*)d_in[4];

  float* out        = (float*)d_out;
  float* out_coarse = out;                                   // [B, E]
  float* out_eid    = out_coarse + (size_t)BATCH * NEXP;     // [B]
  float* out_local  = out_eid + BATCH;                       // [B, C]
  float* out_glob   = out_local + (size_t)BATCH * NCLS;      // [B]

  char* ws = (char*)d_ws;
  constexpr size_t OFF_LIST = 256;
  constexpr size_t OFF_EW   = OFF_LIST + (size_t)NEXP * BATCH * 4;            // list: 1 MB
  constexpr size_t OFF_CW   = OFF_EW + (size_t)NEXP * NCLS * DIM * 2;         // ew_h: 4 MB
  constexpr size_t OFF_FEAT = (OFF_CW + (size_t)NEXP * DIM * 2 + 255) & ~(size_t)255;
  constexpr size_t WS_STAGED = OFF_FEAT + (size_t)BATCH * DIM * 2;            // + 64 MB staged f16 features

  int* counts    = (int*)ws;                       // zeroed every call
  int* list      = (int*)(ws + OFF_LIST);
  _Float16* ew_h = (_Float16*)(ws + OFF_EW);
  _Float16* cw_h = (_Float16*)(ws + OFF_CW);
  const bool staged = (ws_size >= WS_STAGED);
  _Float16* feat_h = staged ? (_Float16*)(ws + OFF_FEAT) : nullptr;

  moe_zero_counts<<<dim3(1), dim3(32), 0, stream>>>(counts);
  {
    int n4 = NEXP * NCLS * DIM / 4;
    moe_cvt_f16<<<dim3((n4 + 255) / 256), dim3(256), 0, stream>>>(expert_w, ew_h, n4);
  }
  {
    int n4 = NEXP * DIM / 4;
    moe_cvt_f16<<<dim3((n4 + 255) / 256), dim3(256), 0, stream>>>(coarse_w, cw_h, n4);
  }
  if (staged) {
    moe_coarse<true><<<dim3(BATCH / 16), dim3(32 * KWAVES), 0, stream>>>(
        feat, cw_h, coarse_b, out_coarse, out_eid, counts, list, feat_h);
    moe_expert<true><<<dim3(BATCH / 16, NEXP), dim3(32 * KWAVES), 0, stream>>>(
        feat, feat_h, ew_h, expert_b, counts, list, out_local, out_glob);
  } else {
    moe_coarse<false><<<dim3(BATCH / 16), dim3(32 * KWAVES), 0, stream>>>(
        feat, cw_h, coarse_b, out_coarse, out_eid, counts, list, nullptr);
    moe_expert<false><<<dim3(BATCH / 16, NEXP), dim3(32 * KWAVES), 0, stream>>>(
        feat, nullptr, ew_h, expert_b, counts, list, out_local, out_glob);
  }
}
